// MODEL_3238405341966
// MI455X (gfx1250) — matrix operand order repaired and checked by reading, not compiled, not run on hardware
//
#include <hip/hip_runtime.h>
#include <hip/hip_bf16.h>
#include <math.h>

typedef __attribute__((ext_vector_type(16))) _Float16 v16h;
typedef __attribute__((ext_vector_type(8)))  _Float16 v8h;
typedef __attribute__((ext_vector_type(8)))  float    v8f;

union V16U { v16h v; v8h p[2]; _Float16 h[16]; };

#define EPS1 1e-6f
#define WD   1e-5f
#define RSQRT2 0.70710678118654752440f

// ---------------------------------------------------------------------------
// small prep kernels
// ---------------------------------------------------------------------------
__global__ void k_zero2(float* p) {
    if (threadIdx.x < 2) p[threadIdx.x] = 0.f;
}

// sum of squares, grid-stride, wave-reduced, atomic accumulate
__global__ void k_l2(const float* __restrict__ w, int n, float* __restrict__ acc) {
    int tid = blockIdx.x * blockDim.x + threadIdx.x;
    int stride = gridDim.x * blockDim.x;
    float s = 0.f;
    for (int i = tid; i < n; i += stride) { float x = w[i]; s += x * x; }
    s += __shfl_xor(s, 1, 32);  s += __shfl_xor(s, 2, 32);
    s += __shfl_xor(s, 4, 32);  s += __shfl_xor(s, 8, 32);
    s += __shfl_xor(s, 16, 32);
    if ((threadIdx.x & 31) == 0) atomicAdd(acc, s);
}

// W [K,N] f32 row-major -> Wt [Npad,K] f16 row-major (zero-padded rows)
__global__ void k_transpose_cvt(const float* __restrict__ src, _Float16* __restrict__ dst,
                                int K, int N, int Npad) {
    int idx = blockIdx.x * blockDim.x + threadIdx.x;
    if (idx >= Npad * K) return;
    int n = idx / K, k = idx - n * K;
    dst[idx] = (n < N) ? (_Float16)src[k * N + n] : (_Float16)0.f;
}

// cov_diag/sqrt + R -> f16 (rows padded 500->512 with zeros)
__global__ void k_cov_prep(const float* __restrict__ R, _Float16* __restrict__ Rh,
                           float* __restrict__ sqrtcov) {
    int c = blockIdx.x * blockDim.x + threadIdx.x;
    if (c >= 512) return;
    if (c < 500) {
        float s = 0.f;
        #pragma unroll 8
        for (int z = 0; z < 64; ++z) { float v = R[c * 64 + z]; s += v * v; Rh[c * 64 + z] = (_Float16)v; }
        sqrtcov[c] = sqrtf(s + 1.f);
    } else {
        for (int z = 0; z < 64; ++z) Rh[c * 64 + z] = (_Float16)0.f;
        sqrtcov[c] = 0.f;
    }
}

// ---------------------------------------------------------------------------
// generic 16x16-tile WMMA GEMM: out(f16) = act(A @ Bt^T + bias)
// A: [M,K] (f32 or f16, row-major); Bt: [N,K] f16 row-major; N%16==0, K%32==0
// one wave per 16x16 tile
// ---------------------------------------------------------------------------
template <bool AF32, bool RELU>
__global__ void k_gemm_wmma(const void* __restrict__ Av, const _Float16* __restrict__ Bt,
                            const float* __restrict__ bias, _Float16* __restrict__ outH,
                            int M, int N, int K) {
    int lane = threadIdx.x & 31;
    int wid  = blockIdx.x * (blockDim.x >> 5) + (threadIdx.x >> 5);
    int tilesN = N >> 4;
    if (wid >= (M >> 4) * tilesN) return;               // wave-uniform
    int mt = wid / tilesN, nt = wid - mt * tilesN;

    int mrow = mt * 16 + (lane & 15);                   // A row (lane & lane+16 same row)
    int n    = nt * 16 + (lane & 15);                   // B/C column for this lane
    int aoff = (lane < 16) ? 0 : 16;                    // A K-chunk split
    int ko   = (lane < 16) ? 0 : 16;                    // B K split

    v8f acc = {};
    for (int kb = 0; kb < K; kb += 32) {
        V16U au;
        if (AF32) {
            const float* ap = (const float*)Av + (size_t)mrow * K + kb + aoff;
            #pragma unroll
            for (int i = 0; i < 8; ++i) { au.h[i] = (_Float16)ap[i]; au.h[8 + i] = (_Float16)ap[8 + i]; }
        } else {
            const _Float16* ap = (const _Float16*)Av + (size_t)mrow * K + kb + aoff;
            au.p[0] = *(const v8h*)ap;
            au.p[1] = *(const v8h*)(ap + 8);
        }
        v16h b = *(const v16h*)(Bt + (size_t)n * K + kb + ko);
        acc = __builtin_amdgcn_wmma_f32_16x16x32_f16(false, au.v, false, b,
                                                     (short)0, acc, false, false);
    }
    float bv = bias[n];
    int ob = mt * 16 + ((lane < 16) ? 0 : 8);
    #pragma unroll
    for (int i = 0; i < 8; ++i) {
        float v = acc[i] + bv;
        if (RELU) v = fmaxf(v, 0.f);
        outH[(size_t)(ob + i) * N + n] = (_Float16)v;
    }
}

// ---------------------------------------------------------------------------
// final trunk layer, fused epilogue: mu -> r_mu, indiv_prob, marginal CE sum
// feat [512,256] f16, Wmut [512,256] f16 (N padded 500->512), K=256
// branchless except the guarded indiv_prob stores.
// ---------------------------------------------------------------------------
__global__ void k_gemm_mu(const _Float16* __restrict__ feat, const _Float16* __restrict__ Wmut,
                          const float* __restrict__ bmu, const float* __restrict__ sqrtcov,
                          const float* __restrict__ Y, float* __restrict__ r_mu,
                          float* __restrict__ indiv, float* __restrict__ ce_acc) {
    const int K = 256;
    int lane = threadIdx.x & 31;
    int wid  = blockIdx.x * (blockDim.x >> 5) + (threadIdx.x >> 5);
    if (wid >= 32 * 32) return;
    int mt = wid >> 5, nt = wid & 31;

    int mrow = mt * 16 + (lane & 15);
    int n    = nt * 16 + (lane & 15);
    int aoff = (lane < 16) ? 0 : 16;
    int ko   = (lane < 16) ? 0 : 16;

    v8f acc = {};
    #pragma unroll 2
    for (int kb = 0; kb < K; kb += 32) {
        V16U au;
        const _Float16* ap = feat + (size_t)mrow * K + kb + aoff;
        au.p[0] = *(const v8h*)ap;
        au.p[1] = *(const v8h*)(ap + 8);
        v16h b = *(const v16h*)(Wmut + (size_t)n * K + kb + ko);
        acc = __builtin_amdgcn_wmma_f32_16x16x32_f16(false, au.v, false, b,
                                                     (short)0, acc, false, false);
    }
    bool  cv   = (n < 500);
    int   nc   = cv ? n : 499;                 // clamped -> always-valid loads
    float mask = cv ? 1.f : 0.f;
    float bv = bmu[nc];
    float sc = sqrtcov[nc] * mask;             // pad columns -> r_mu = 0
    int ob = mt * 16 + ((lane < 16) ? 0 : 8);
    float ce = 0.f;
    float pv[8];
    #pragma unroll
    for (int i = 0; i < 8; ++i) {
        int brow = ob + i;
        float mu = acc[i] + bv;
        r_mu[(size_t)brow * 512 + n] = mu * sc;
        float p = 1.f / (1.f + __expf(-1.70169f * mu));
        p = p * (1.f - EPS1) + 0.5f * EPS1;
        pv[i] = p;
        float y = Y[(size_t)brow * 500 + nc];
        ce += mask * (y * __logf(p) + (1.f - y) * __logf(1.f - p));
    }
    if (cv) {
        #pragma unroll
        for (int i = 0; i < 8; ++i) indiv[(size_t)(ob + i) * 500 + n] = pv[i];
    }
    ce += __shfl_xor(ce, 1, 32);  ce += __shfl_xor(ce, 2, 32);
    ce += __shfl_xor(ce, 4, 32);  ce += __shfl_xor(ce, 8, 32);
    ce += __shfl_xor(ce, 16, 32);
    if (lane == 0) atomicAdd(ce_acc, ce);
}

// ---------------------------------------------------------------------------
// fused MC-sample kernel: sample_r = noise @ R^T + r_mu (WMMA), then
// probit CDF + bernoulli log-likelihood, reduced over C -> logprob[s,b]
// one wave = 16 rows (one s, 16 b's) x all 500 classes. 3200 waves.
// fully branchless epilogue (clamped index + 0/1 mask).
// ---------------------------------------------------------------------------
__global__ void k_sample_nll(const float* __restrict__ noise, const _Float16* __restrict__ Rh,
                             const float* __restrict__ r_mu, const float* __restrict__ Y,
                             float* __restrict__ logprob) {
    int lane = threadIdx.x & 31;
    int wid  = blockIdx.x * (blockDim.x >> 5) + (threadIdx.x >> 5);
    if (wid >= 3200) return;                            // wave-uniform
    int s  = wid >> 5;                                  // sample index
    int bt = wid & 31;                                  // 16-row b tile

    // A tile: 16x64 of noise[s, bt*16 .. +15, :], f32 -> f16 in registers once
    int mrow = lane & 15;
    const float* ap = noise + ((size_t)(s * 512 + bt * 16 + mrow)) * 64 + ((lane < 16) ? 0 : 16);
    V16U a0, a1;
    #pragma unroll
    for (int i = 0; i < 8; ++i) {
        a0.h[i]     = (_Float16)ap[i];
        a0.h[8 + i] = (_Float16)ap[8 + i];
        a1.h[i]     = (_Float16)ap[32 + i];
        a1.h[8 + i] = (_Float16)ap[40 + i];
    }

    int ko    = (lane < 16) ? 0 : 16;
    int mbase = bt * 16 + ((lane < 16) ? 0 : 8);
    float rs[8];
    #pragma unroll
    for (int i = 0; i < 8; ++i) rs[i] = 0.f;

    for (int nt = 0; nt < 32; ++nt) {
        int c = nt * 16 + (lane & 15);
        const _Float16* bp = Rh + (size_t)c * 64 + ko;
        v16h b0 = *(const v16h*)bp;
        v16h b1 = *(const v16h*)(bp + 32);
        v8f acc0 = {}, acc1 = {};                       // independent WMMAs, no D->C chain
        acc0 = __builtin_amdgcn_wmma_f32_16x16x32_f16(false, a0.v, false, b0,
                                                      (short)0, acc0, false, false);
        acc1 = __builtin_amdgcn_wmma_f32_16x16x32_f16(false, a1.v, false, b1,
                                                      (short)0, acc1, false, false);
        bool  cv   = (c < 500);
        int   cc   = cv ? c : 499;                      // clamp -> loads always valid
        float mask = cv ? 1.f : 0.f;
        #pragma unroll
        for (int i = 0; i < 8; ++i) {
            int brow = mbase + i;
            float x = acc0[i] + acc1[i] + r_mu[(size_t)brow * 512 + cc];
            float t = x * RSQRT2;
            // E = ndtr(x)*(1-eps)+eps/2; 1-E via erfc symmetry (accurate in both tails)
            float E  = 0.5f * erfcf(-t) * (1.f - EPS1) + 0.5f * EPS1;
            float om = 0.5f * erfcf( t) * (1.f - EPS1) + 0.5f * EPS1;
            float y = Y[(size_t)brow * 500 + cc];
            rs[i] += mask * (y * __logf(E) + (1.f - y) * __logf(om));
        }
    }
    // reduce each row over the 16 lanes holding its columns
    #pragma unroll
    for (int i = 0; i < 8; ++i) {
        float v = rs[i];
        v += __shfl_xor(v, 1, 32); v += __shfl_xor(v, 2, 32);
        v += __shfl_xor(v, 4, 32); v += __shfl_xor(v, 8, 32);
        rs[i] = v;
    }
    if ((lane & 15) == 0) {
        int bb = bt * 16 + ((lane < 16) ? 0 : 8);
        #pragma unroll
        for (int i = 0; i < 8; ++i) logprob[(size_t)s * 512 + bb + i] = rs[i];
    }
}

// ---------------------------------------------------------------------------
// log-sum-exp over S per b, then mean -> nll; finalize all scalars
// ---------------------------------------------------------------------------
__global__ void k_finalize(const float* __restrict__ logprob, const float* __restrict__ scal,
                           float* __restrict__ out) {
    __shared__ float red[512];
    int b = threadIdx.x;
    float mx = -3.402823466e38f;
    for (int s = 0; s < 100; ++s) mx = fmaxf(mx, logprob[s * 512 + b]);
    float se = 0.f;
    for (int s = 0; s < 100; ++s) se += __expf(logprob[s * 512 + b] - mx);
    red[b] = -__logf(se * 0.01f) - mx;                  // nll per batch element
    __syncthreads();
    for (int off = 256; off > 0; off >>= 1) {
        if (b < off) red[b] += red[b + off];
        __syncthreads();
    }
    if (b == 0) {
        float nll  = red[0] / 512.f;
        float marg = -scal[1] / 512.f;
        float l2   = WD * scal[0];
        out[256000] = nll;
        out[256001] = marg;
        out[256002] = l2;
        out[256003] = l2 + nll;
    }
}

// ---------------------------------------------------------------------------
extern "C" void kernel_launch(void* const* d_in, const int* in_sizes, int n_in,
                              void* d_out, int out_size, void* d_ws, size_t ws_size,
                              hipStream_t stream) {
    const float* X     = (const float*)d_in[0];   // [512,512]
    const float* Y     = (const float*)d_in[1];   // [512,500]
    const float* noise = (const float*)d_in[2];   // [100,512,64]
    const float* W1    = (const float*)d_in[3];   // [512,128]
    const float* b1    = (const float*)d_in[4];
    const float* W2    = (const float*)d_in[5];   // [128,256]
    const float* b2    = (const float*)d_in[6];
    const float* W3    = (const float*)d_in[7];   // [256,256]
    const float* b3    = (const float*)d_in[8];
    const float* Wmu   = (const float*)d_in[9];   // [256,500]
    const float* bmu   = (const float*)d_in[10];
    const float* R     = (const float*)d_in[11];  // [500,64]
    float* out = (float*)d_out;

    char* base = (char*)d_ws;
    size_t off = 0;
    auto alloc = [&](size_t bytes) { size_t r = off; off = (off + bytes + 255) & ~(size_t)255; return r; };

    float*     scal    = (float*)(base + alloc(2 * sizeof(float)));     // [l2sq, ce]
    float*     sqrtcov = (float*)(base + alloc(512 * 4));
    _Float16*  Rh      = (_Float16*)(base + alloc(512 * 64 * 2));
    _Float16*  W1t     = (_Float16*)(base + alloc(128 * 512 * 2));
    _Float16*  W2t     = (_Float16*)(base + alloc(256 * 128 * 2));
    _Float16*  W3t     = (_Float16*)(base + alloc(256 * 256 * 2));
    _Float16*  Wmut    = (_Float16*)(base + alloc(512 * 256 * 2));
    _Float16*  h1      = (_Float16*)(base + alloc(512 * 128 * 2));
    _Float16*  h2      = (_Float16*)(base + alloc(512 * 256 * 2));
    _Float16*  feat    = (_Float16*)(base + alloc(512 * 256 * 2));
    float*     r_mu    = (float*)(base + alloc(512 * 512 * 4));
    float*     logprob = (float*)(base + alloc(100 * 512 * 4));

    // ---- prep ----
    k_zero2<<<1, 64, 0, stream>>>(scal);
    k_l2<<<64, 256, 0, stream>>>(W1, 512 * 128, scal);
    k_l2<<<64, 256, 0, stream>>>(W2, 128 * 256, scal);
    k_l2<<<64, 256, 0, stream>>>(W3, 256 * 256, scal);
    k_l2<<<64, 256, 0, stream>>>(Wmu, 256 * 500, scal);
    k_transpose_cvt<<<(128 * 512 + 255) / 256, 256, 0, stream>>>(W1, W1t, 512, 128, 128);
    k_transpose_cvt<<<(256 * 128 + 255) / 256, 256, 0, stream>>>(W2, W2t, 128, 256, 256);
    k_transpose_cvt<<<(256 * 256 + 255) / 256, 256, 0, stream>>>(W3, W3t, 256, 256, 256);
    k_transpose_cvt<<<(512 * 256 + 255) / 256, 256, 0, stream>>>(Wmu, Wmut, 256, 500, 512);
    k_cov_prep<<<2, 256, 0, stream>>>(R, Rh, sqrtcov);

    // ---- MLP trunk (WMMA) ----
    k_gemm_wmma<true,  true><<<64, 128, 0, stream>>>(X,   W1t, b1, h1,   512, 128, 512);
    k_gemm_wmma<false, true><<<128, 128, 0, stream>>>(h1, W2t, b2, h2,   512, 256, 128);
    k_gemm_wmma<false, true><<<128, 128, 0, stream>>>(h2, W3t, b3, feat, 512, 256, 256);
    k_gemm_mu<<<256, 128, 0, stream>>>(feat, Wmut, bmu, sqrtcov, Y, r_mu, out, scal + 1);

    // ---- fused MC sampling + probit NLL: 3200 waves ----
    k_sample_nll<<<800, 128, 0, stream>>>(noise, Rh, r_mu, Y, logprob);

    // ---- scalars ----
    k_finalize<<<1, 512, 0, stream>>>(logprob, scal, out);

    (void)in_sizes; (void)n_in; (void)out_size; (void)ws_size;
}
